// AutoregLSTM_6828998000900
// MI455X (gfx1250) — matrix operand order repaired and checked by reading, not compiled, not run on hardware
//
#include <hip/hip_runtime.h>

// ---------------------------------------------------------------------------
// Autoregressive LSTM for MI455X (gfx1250, wave32, WMMA).
// 16 persistent workgroups (one per 16-row batch slice), no inter-WG sync.
// All GEMMs via v_wmma_f32_16x16x32_bf16; cell state lives in VGPRs.
// Weights stream from L2 every step (2.9 MB bf16 total, 192 MB L2): the
// compiler must NOT hoist them out of the time loop (no restrict + asm
// memory barrier per step), or it spills 700+ VGPRs to scratch.
// Accumulators are zero-initialized (rematerializable) and bias is folded
// into the gate epilogue, so nothing loop-invariant spills to scratch.
// ---------------------------------------------------------------------------

typedef __bf16 v16bf  __attribute__((ext_vector_type(16)));
typedef __bf16 bf16x8 __attribute__((ext_vector_type(8)));
typedef float  v8f    __attribute__((ext_vector_type(8)));

#define B_   256
#define T_   1024
#define F_   128
#define H_   512
#define D_   64
#define FD_  192          // F + D (concat [x_t | p])
#define INP_S 200         // padded LDS row stride (bf16) -> bank-conflict-free b128 reads
#define H_S   520         // padded LDS row stride (bf16) for h

// workspace layout (bf16, transposed weights: [N][K] so B-fragments are contiguous)
#define WIT_OFF 0
#define WHT_OFF (2048*192*2)            //   786432 B
#define WOT_OFF (WHT_OFF + 2048*512*2)  //  2883584 B
#define WS_NEEDED (WOT_OFF + 64*512*2)  //  2949120 B total

// ---- prologue: convert f32 weights to transposed bf16 (runs every call) ----
__global__ void wconvert(const float* __restrict__ Wi, const float* __restrict__ Wh,
                         const float* __restrict__ Wo,
                         __bf16* __restrict__ WiT, __bf16* __restrict__ WhT,
                         __bf16* __restrict__ WoT) {
  int idx = blockIdx.x * blockDim.x + threadIdx.x;
  int stride = gridDim.x * blockDim.x;
  for (int i = idx; i < 2048 * FD_; i += stride) {        // WiT[n][k] = Wi[k][n]
    int n = i / FD_, k = i - n * FD_;
    WiT[i] = (__bf16)Wi[k * 2048 + n];
  }
  for (int i = idx; i < 2048 * H_; i += stride) {         // WhT[n][k] = Wh[k][n]
    int n = i >> 9, k = i & (H_ - 1);
    WhT[i] = (__bf16)Wh[k * 2048 + n];
  }
  for (int i = idx; i < D_ * H_; i += stride) {           // WoT[n][k] = Wo[k][n]
    int n = i >> 9, k = i & (H_ - 1);
    WoT[i] = (__bf16)Wo[k * D_ + n];
  }
}

// ---- fast activations (guarded against exp overflow) ----
__device__ __forceinline__ float sigm(float x) { return 1.0f / (1.0f + __expf(-x)); }
__device__ __forceinline__ float tanh_fast(float x) {
  float a = fabsf(x);
  float e = __expf(-2.0f * a);
  float r = (1.0f - e) / (1.0f + e);
  return copysignf(r, x);
}

// A-fragment (16x32 bf16, ISA 7.12.2): lane row = lane&15, half = lane>>4.
// elems 0..15 <-> K = kbase + 16*half + 0..15 (matches B-fragment order)
__device__ __forceinline__ v16bf load_a_frag(const __bf16* row, int kbase, int lh) {
  union { bf16x8 h[2]; v16bf v; } u;
  u.h[0] = *(const bf16x8*)(row + kbase + lh * 16);
  u.h[1] = *(const bf16x8*)(row + kbase + lh * 16 + 8);
  return u.v;
}
// B-fragment (32x16 bf16): lane col = lane&15; elems <-> K = kbase + 16*half + 0..15.
// Weights stored transposed [N][K] -> one contiguous 32-byte load per lane.
// `base` already points at this lane's column row (n*strideK).
__device__ __forceinline__ v16bf load_b_frag(const __bf16* base, int kbase, int lh) {
  return *(const v16bf*)(base + kbase + lh * 16);
}

#define WMMA_BF16(A, Bf, C) \
  __builtin_amdgcn_wmma_f32_16x16x32_bf16(false, (A), false, (Bf), (short)0, (C), false, false)

// ---------------------------------------------------------------------------
__launch_bounds__(1024, 1)
__global__ void lstm_persist(const float* __restrict__ c0, const float* __restrict__ h0,
                             const float* __restrict__ p0, const float* __restrict__ x,
                             const float* __restrict__ bv, const float* __restrict__ bo,
                             const __bf16* WiT,   // no restrict: must be re-loaded per step
                             const __bf16* WhT,
                             const __bf16* WoT,
                             float* out) {
  __shared__ __align__(16) __bf16 inp[16 * INP_S];   // [x_t | p] bf16, padded rows
  __shared__ __align__(16) __bf16 hbuf[16 * H_S];    // h_t bf16, padded rows
  __shared__ __align__(16) float  pbuf[16 * D_];     // p_t (== y_{t-1}) f32

  const int tid  = threadIdx.x;
  const int w    = tid >> 5;          // wave 0..31 -> hidden slice
  const int ln   = tid & 15;          // lane row/col within tile
  const int lh   = (tid >> 4) & 1;    // lane half
  const int m0   = blockIdx.x * 16;   // batch rows [m0, m0+16)
  const int hid0 = w * 16;

  float* out_c  = out;
  float* out_h  = out + B_ * H_;
  float* out_p  = out + 2 * B_ * H_;
  float* out_ys = out + 2 * B_ * H_ + B_ * D_;

  // persistent cell-state tile (C/D layout: elem r <-> row r+8*lh, col hid0+ln)
  float creg[8];
#pragma unroll
  for (int r = 0; r < 8; ++r)
    creg[r] = c0[(size_t)(m0 + r + 8 * lh) * H_ + hid0 + ln];

  // per-lane bias scalars (folded into the gate epilogue, NOT into the
  // accumulator init, so accumulator init is a rematerializable zero).
  const float bias_i = bv[0 * H_ + hid0 + ln];
  const float bias_f = bv[1 * H_ + hid0 + ln];
  const float bias_g = bv[2 * H_ + hid0 + ln];
  const float bias_o = bv[3 * H_ + hid0 + ln];
  const float bias_y = bo[(w & 3) * 16 + ln];

  // per-wave weight base pointers (one column-row per gate tile)
  const __bf16* bWi[4];
  const __bf16* bWh[4];
#pragma unroll
  for (int g = 0; g < 4; ++g) {
    bWi[g] = WiT + (size_t)(g * H_ + hid0 + ln) * FD_;
    bWh[g] = WhT + (size_t)(g * H_ + hid0 + ln) * H_;
  }
  const __bf16* bWo = WoT + (size_t)((w & 3) * 16 + ln) * H_;

  for (int i = tid; i < 16 * H_; i += 1024) {
    int rr = i >> 9, cc = i & (H_ - 1);
    hbuf[rr * H_S + cc] = (__bf16)h0[(size_t)(m0 + rr) * H_ + cc];
  }
  for (int i = tid; i < 16 * D_; i += 1024)
    pbuf[i] = p0[(size_t)m0 * D_ + i];
  __syncthreads();

  for (int t = 0; t < T_; ++t) {
    // Compiler barrier: weight loads below must stay inside the time loop
    // (streamed from L2 each step) instead of being hoisted + spilled.
    asm volatile("" ::: "memory");

    // ---- stage 1: build inp = bf16([x_t | p]); prefetch next x tile
    for (int i = tid; i < 16 * FD_; i += 1024) {
      int rr = i / FD_;
      int cc = i - rr * FD_;
      float v = (cc < F_) ? x[(size_t)(m0 + rr) * T_ * F_ + (size_t)t * F_ + cc]
                          : pbuf[rr * D_ + (cc - F_)];
      inp[rr * INP_S + cc] = (__bf16)v;
    }
    if (t + 1 < T_ && tid < 64) {      // global_prefetch_b8: next step's x rows
      int rr = tid >> 2, ch = tid & 3;
      __builtin_prefetch(&x[(size_t)(m0 + rr) * T_ * F_ + (size_t)(t + 1) * F_ + ch * 32], 0, 1);
    }
    __syncthreads();

    // ---- stage 2: z = inp@Wi + h@Wh ; four gate tiles (i,f,g,o) per wave
    v8f zi = (v8f)(0.0f), zf = (v8f)(0.0f), zg = (v8f)(0.0f), zo = (v8f)(0.0f);

    const __bf16* arow = inp + ln * INP_S;
#pragma unroll
    for (int k = 0; k < FD_; k += 32) {
      // issue all loads first so waits can be partial / overlapped with WMMAs
      v16bf a  = load_a_frag(arow, k, lh);
      v16bf b0 = load_b_frag(bWi[0], k, lh);
      v16bf b1 = load_b_frag(bWi[1], k, lh);
      v16bf b2 = load_b_frag(bWi[2], k, lh);
      v16bf b3 = load_b_frag(bWi[3], k, lh);
      zi = WMMA_BF16(a, b0, zi);
      zf = WMMA_BF16(a, b1, zf);
      zg = WMMA_BF16(a, b2, zg);
      zo = WMMA_BF16(a, b3, zo);
    }
    const __bf16* hrow = hbuf + ln * H_S;
#pragma unroll 4
    for (int k = 0; k < H_; k += 32) {
      v16bf a  = load_a_frag(hrow, k, lh);
      v16bf b0 = load_b_frag(bWh[0], k, lh);
      v16bf b1 = load_b_frag(bWh[1], k, lh);
      v16bf b2 = load_b_frag(bWh[2], k, lh);
      v16bf b3 = load_b_frag(bWh[3], k, lh);
      zi = WMMA_BF16(a, b0, zi);
      zf = WMMA_BF16(a, b1, zf);
      zg = WMMA_BF16(a, b2, zg);
      zo = WMMA_BF16(a, b3, zo);
    }

    float nh[8];
#pragma unroll
    for (int r = 0; r < 8; ++r) {
      float nc = sigm(zf[r] + bias_f) * creg[r]
               + sigm(zi[r] + bias_i) * tanh_fast(zg[r] + bias_g);
      creg[r] = nc;
      nh[r]   = sigm(zo[r] + bias_o) * tanh_fast(nc);
    }
    __syncthreads();            // all waves done reading hbuf

    // ---- stage 3: publish new h (bf16)
#pragma unroll
    for (int r = 0; r < 8; ++r)
      hbuf[(r + 8 * lh) * H_S + hid0 + ln] = (__bf16)nh[r];
    if (t == T_ - 1) {
#pragma unroll
      for (int r = 0; r < 8; ++r) {
        size_t row = (size_t)(m0 + r + 8 * lh);
        out_c[row * H_ + hid0 + ln] = creg[r];
        out_h[row * H_ + hid0 + ln] = nh[r];
      }
    }
    __syncthreads();            // hbuf ready

    // ---- stage 4: y = new_h @ Wo + bo (waves 0..3, one 16x16 tile each)
    if (w < 4) {
      const int n0 = w * 16;
      v8f ya = (v8f)(0.0f);
      const __bf16* hr = hbuf + ln * H_S;
#pragma unroll 4
      for (int k = 0; k < H_; k += 32) {
        v16bf a = load_a_frag(hr, k, lh);
        v16bf b = load_b_frag(bWo, k, lh);
        ya = WMMA_BF16(a, b, ya);
      }
#pragma unroll
      for (int r = 0; r < 8; ++r) {
        int rl = r + 8 * lh;
        float yv = ya[r] + bias_y;
        pbuf[rl * D_ + n0 + ln] = yv;
        out_ys[(size_t)(m0 + rl) * T_ * D_ + (size_t)t * D_ + n0 + ln] = yv;
        if (t == T_ - 1) out_p[(size_t)(m0 + rl) * D_ + n0 + ln] = yv;
      }
    }
    __syncthreads();            // pbuf ready for next t
  }
}

// ---------------------------------------------------------------------------
extern "C" void kernel_launch(void* const* d_in, const int* in_sizes, int n_in,
                              void* d_out, int out_size, void* d_ws, size_t ws_size,
                              hipStream_t stream) {
  (void)in_sizes; (void)n_in; (void)out_size; (void)ws_size; // needs WS_NEEDED (~2.95 MB)

  const float* c0 = (const float*)d_in[0];
  const float* h0 = (const float*)d_in[1];
  const float* p0 = (const float*)d_in[2];
  const float* x  = (const float*)d_in[3];
  const float* Wi = (const float*)d_in[4];
  const float* Wh = (const float*)d_in[5];
  const float* bv = (const float*)d_in[6];
  const float* Wo = (const float*)d_in[7];
  const float* bo = (const float*)d_in[8];

  __bf16* WiT = (__bf16*)((char*)d_ws + WIT_OFF);
  __bf16* WhT = (__bf16*)((char*)d_ws + WHT_OFF);
  __bf16* WoT = (__bf16*)((char*)d_ws + WOT_OFF);

  wconvert<<<512, 256, 0, stream>>>(Wi, Wh, Wo, WiT, WhT, WoT);
  lstm_persist<<<B_ / 16, 1024, 0, stream>>>(c0, h0, p0, x, bv, bo, WiT, WhT, WoT,
                                             (float*)d_out);
}
